// MetaoptHead_84877143704064
// MI455X (gfx1250) — matrix operand order repaired and checked by reading, not compiled, not run on hardware
//
#include <hip/hip_runtime.h>
#include <hip/hip_bf16.h>

typedef __attribute__((ext_vector_type(16))) _Float16 v16h;
typedef __attribute__((ext_vector_type(8)))  _Float16 v8h;
typedef __attribute__((ext_vector_type(8)))  float    v8f;

#define NWAY 5
#define NS   25
#define NZ   125
#define NQ   75
#define DF   640
#define NSP  32     // padded train rows per batch
#define NQP  80     // padded query rows per batch
#define BETA_C 1.3f
#define C_REG  0.1f

// ---------------- Kernel 1: simple_transform + l2norm, emit f16 (padded rows = 0) ----------------
__device__ __forceinline__ float simple_tf(float x) {
  float xp = fmaxf(x, 0.f);
  float xn = fminf(x, 0.f);
  float pos =  powf(logf(1.f / (xp + 1e-5f) + 1.f), -BETA_C);
  float neg = -powf(logf(1.f / (-xn + 1e-5f) + 1.f), -BETA_C);
  return pos + neg;
}

__global__ void __launch_bounds__(128) k_prep(const float* __restrict__ ftest,
                                              const float* __restrict__ ftrain,
                                              _Float16* __restrict__ fqn,
                                              _Float16* __restrict__ ftn, int B) {
  __shared__ float buf[DF];
  __shared__ float part[128];
  int rid = blockIdx.x;                  // B * (NQP + NSP) blocks
  int b = rid / (NQP + NSP);
  int r = rid % (NQP + NSP);
  int tid = threadIdx.x;
  const float* src = nullptr;
  _Float16* dst;
  if (r < NQP) {
    dst = fqn + ((size_t)b * NQP + r) * DF;
    if (r < NQ) src = ftest + ((size_t)b * NQ + r) * DF;
  } else {
    int r2 = r - NQP;
    dst = ftn + ((size_t)b * NSP + r2) * DF;
    if (r2 < NS) src = ftrain + ((size_t)b * NS + r2) * DF;
  }
  if (src == nullptr) {                  // zero pad row
    for (int i = tid; i < DF; i += blockDim.x) dst[i] = (_Float16)0.f;
    return;
  }
  float ss = 0.f;
  for (int i = tid; i < DF; i += blockDim.x) {
    float v = simple_tf(src[i]);
    buf[i] = v;
    ss += v * v;
  }
  part[tid] = ss;
  __syncthreads();
  for (int off = 64; off > 0; off >>= 1) {
    if (tid < off) part[tid] += part[tid + off];
    __syncthreads();
  }
  float rn = 1.f / fmaxf(sqrtf(part[0]), 1e-12f);
  for (int i = tid; i < DF; i += blockDim.x) dst[i] = (_Float16)(buf[i] * rn);
}

// ---------------- WMMA 16x16 tile GEMM: D = A(rows x 640) * Bsrc(rows x 640)^T ----------------
// f16 WMMA fragments as loaded here, contiguous k = 16h + s (wave32), rows pre-padded with zeros so no masking:
//  A 16x32:  lane (h,r): loads halfwords [k0+16h, +16)                   -> two b128 loads
//  B 32x16:  lane (h,r): loads halfwords [k0+16h, +16)                   -> two b128 loads
//  C/D f32:  lane col n = r, vgpr v: row = v + 8h
__device__ __forceinline__ v8f wmma_tile(const _Float16* __restrict__ A,
                                         const _Float16* __restrict__ Bsrc,
                                         int m0, int n0) {
  int lane = threadIdx.x & 31;
  int h = lane >> 4, r = lane & 15;
  const _Float16* Ap = A    + (size_t)(m0 + r) * DF;
  const _Float16* Bp = Bsrc + (size_t)(n0 + r) * DF;
  v8f acc0 = {0.f, 0.f, 0.f, 0.f, 0.f, 0.f, 0.f, 0.f};
  v8f acc1 = {0.f, 0.f, 0.f, 0.f, 0.f, 0.f, 0.f, 0.f};
#pragma unroll 2
  for (int k0 = 0; k0 < DF; k0 += 64) {
    {
      v16h bv = *(const v16h*)(Bp + k0 + 16 * h);
      v16h av = *(const v16h*)(Ap + k0 + 16 * h);
      acc0 = __builtin_amdgcn_wmma_f32_16x16x32_f16(false, av, false, bv, (short)0,
                                                    acc0, false, false);
    }
    {
      v16h bv = *(const v16h*)(Bp + k0 + 32 + 16 * h);
      v16h av = *(const v16h*)(Ap + k0 + 32 + 16 * h);
      acc1 = __builtin_amdgcn_wmma_f32_16x16x32_f16(false, av, false, bv, (short)0,
                                                    acc1, false, false);
    }
  }
#pragma unroll
  for (int e = 0; e < 8; ++e) acc0[e] += acc1[e];
  return acc0;
}

// ---------------- Kernel 2: K = ft ft^T (padded 32x32, stride 32), 4 waves = 2x2 tiles ----------------
__global__ void __launch_bounds__(128) k_gram(const _Float16* __restrict__ ftn,
                                              float* __restrict__ Kb, int B) {
  int b = blockIdx.x;
  const _Float16* F = ftn + (size_t)b * NSP * DF;
  int wid = threadIdx.x >> 5;      // 0..3
  int mi = wid >> 1, ni = wid & 1;
  v8f acc = wmma_tile(F, F, mi * 16, ni * 16);
  int lane = threadIdx.x & 31;
  int h = lane >> 4, r = lane & 15;
  float* outp = Kb + (size_t)b * NSP * NSP;
#pragma unroll
  for (int v = 0; v < 8; ++v) {
    int mm = mi * 16 + v + 8 * h, nn = ni * 16 + r;
    outp[mm * NSP + nn] = acc[v];                      // unconditional (padded)
  }
}

// ---------------- Kernel 3: compat = ft fq^T (padded 32x80, stride 80), 10 waves = 2x5 tiles ----------------
__global__ void __launch_bounds__(320) k_compat(const _Float16* __restrict__ ftn,
                                                const _Float16* __restrict__ fqn,
                                                float* __restrict__ compat, int B) {
  int b = blockIdx.x;
  const _Float16* F  = ftn + (size_t)b * NSP * DF;
  const _Float16* Qf = fqn + (size_t)b * NQP * DF;
  int wid = threadIdx.x >> 5;      // 0..9
  int mi = wid / 5, ni = wid % 5;
  v8f acc = wmma_tile(F, Qf, mi * 16, ni * 16);
  int lane = threadIdx.x & 31;
  int h = lane >> 4, r = lane & 15;
  float* outp = compat + (size_t)b * NSP * NQP;
#pragma unroll
  for (int v = 0; v < 8; ++v) {
    int mm = mi * 16 + v + 8 * h, nn = ni * 16 + r;
    outp[mm * NQP + nn] = acc[v];                      // unconditional (padded)
  }
}

// ---------------- Kernel 4: structure-exploited interior-point QP ----------------
// Q = K (x) I5 + I125  -> per-class block M = K + I25 (all classes identical).
// KKT Schur system reduces to five 25x25 SPD solves + one 25x25 general solve.
struct QPS {
  float Mm[625], Lm[625], Minv[625], Wm[625], Wk[625];
  float NL[5][625], Gm[5][625];
  float tv[125], iQr[125], h2[125], h1[25], fac[25], wv[25];
  float xv[125], sv[125], zv[125], yv[25];
  float rxv[125], rsv[125], rzv[125], ryv[25];
  float dvec[125], rsc[125];
  float dxa[125], dsa[125], dza[125], dya[25];
  float dxc[125], dsc[125], dzc[125], dyc[25];
  float bestx[125], zer125[125], zer25[25];
  float red[8];
};

__device__ void chol_multi(float* A, int nmat, int tid, int nthr) {
  // nmat contiguous 25x25 matrices, in-place lower Cholesky
  for (int j = 0; j < 25; ++j) {
    if (tid < nmat) {
      float dv = A[tid * 625 + j * 26];
      A[tid * 625 + j * 26] = sqrtf(fmaxf(dv, 1e-20f));
    }
    __syncthreads();
    for (int idx = tid; idx < nmat * 25; idx += nthr) {
      int m = idx / 25, i = idx % 25;
      if (i > j) A[m * 625 + i * 25 + j] /= A[m * 625 + j * 26];
    }
    __syncthreads();
    for (int idx = tid; idx < nmat * 625; idx += nthr) {
      int m = idx / 625, rem = idx % 625, i = rem / 25, k = rem % 25;
      if (i > j && k > j && k <= i)
        A[m * 625 + rem] -= A[m * 625 + i * 25 + j] * A[m * 625 + k * 25 + j];
    }
    __syncthreads();
  }
}

__device__ __forceinline__ float step_min125(const float* v, const float* dv) {
  float m = 1e12f;
  for (int i = 0; i < 125; ++i) {
    float a = (dv[i] < 0.f) ? (-v[i] / dv[i]) : 1e12f;
    m = fminf(m, a);
  }
  return m;
}

__device__ void solve_kkt(QPS& S, const float* d, const float* rx, const float* rs,
                          const float* rz, const float* ry,
                          float* dx, float* ds, float* dz, float* dy,
                          bool refactor, int tid, int nthr) {
  // iQrx = Q^-1 rx  (per class: Minv @ rx_col)
  for (int idx = tid; idx < 125; idx += nthr) {
    int i = idx / 5, k = idx % 5;
    float acc = 0.f;
    for (int j = 0; j < 25; ++j) acc += S.Minv[i * 25 + j] * rx[j * 5 + k];
    S.iQr[idx] = acc;
  }
  __syncthreads();
  for (int e = tid; e < 25; e += nthr) {
    float acc = 0.f;
    for (int k = 0; k < 5; ++k) acc += S.iQr[e * 5 + k];
    S.h1[e] = acc - ry[e];
  }
  for (int idx = tid; idx < 125; idx += nthr)
    S.h2[idx] = S.iQr[idx] + rs[idx] / d[idx] - rz[idx];
  __syncthreads();
  if (refactor) {
    for (int idx = tid; idx < 5 * 625; idx += nthr) {
      int m = idx / 625, rem = idx % 625;
      S.NL[m][rem] = S.Minv[rem];
    }
    __syncthreads();
    for (int idx = tid; idx < 125; idx += nthr) {
      int i = idx / 5, k = idx % 5;
      S.NL[k][i * 26] = S.Minv[i * 26] + 1.f / d[idx];
    }
    __syncthreads();
    chol_multi(&S.NL[0][0], 5, tid, nthr);
    // G_k = N_k^-1 Minv  (125 columns, one per lane)
    if (tid < 125) {
      int m = tid / 25, c = tid % 25;
      const float* L = S.NL[m];
      float v[25];
      for (int i = 0; i < 25; ++i) {
        float acc = S.Minv[i * 25 + c];
        for (int j = 0; j < i; ++j) acc -= L[i * 25 + j] * v[j];
        v[i] = acc / L[i * 26];
      }
      for (int i = 24; i >= 0; --i) {
        float acc = v[i];
        for (int j = i + 1; j < 25; ++j) acc -= L[j * 25 + i] * v[j];
        v[i] = acc / L[i * 26];
      }
      for (int i = 0; i < 25; ++i) S.Gm[m][i * 25 + c] = v[i];
    }
    __syncthreads();
    // W = 5I - sum_k G_k
    for (int idx = tid; idx < 625; idx += nthr) {
      int i = idx / 25, j = idx % 25;
      float acc = (i == j) ? 5.f : 0.f;
      for (int m = 0; m < 5; ++m) acc -= S.Gm[m][idx];
      S.Wm[idx] = acc;
    }
    __syncthreads();
  }
  // t_k = N_k^-1 h2_k  (5 columns)
  if (tid < 5) {
    const float* L = S.NL[tid];
    float v[25];
    for (int i = 0; i < 25; ++i) {
      float acc = S.h2[i * 5 + tid];
      for (int j = 0; j < i; ++j) acc -= L[i * 25 + j] * v[j];
      v[i] = acc / L[i * 26];
    }
    for (int i = 24; i >= 0; --i) {
      float acc = v[i];
      for (int j = i + 1; j < 25; ++j) acc -= L[j * 25 + i] * v[j];
      v[i] = acc / L[i * 26];
    }
    for (int i = 0; i < 25; ++i) S.tv[tid * 25 + i] = v[i];
  }
  __syncthreads();
  // rhs = M h1 - sum_k t_k ; Wk = W (work copy for elimination)
  if (tid < 25) {
    float acc = 0.f;
    for (int f = 0; f < 25; ++f) acc += S.Mm[tid * 25 + f] * S.h1[f];
    for (int m = 0; m < 5; ++m) acc -= S.tv[m * 25 + tid];
    S.wv[tid] = acc;
  }
  for (int idx = tid; idx < 625; idx += nthr) S.Wk[idx] = S.Wm[idx];
  __syncthreads();
  // Gaussian elimination (no pivot): Wk * dy' = wv
  for (int p = 0; p < 25; ++p) {
    if (tid > p && tid < 25) S.fac[tid] = S.Wk[tid * 25 + p] / S.Wk[p * 26];
    __syncthreads();
    for (int idx = tid; idx < 625; idx += nthr) {
      int i = idx / 25, j = idx % 25;
      if (i > p && j >= p) S.Wk[idx] -= S.fac[i] * S.Wk[p * 25 + j];
    }
    if (tid > p && tid < 25) S.wv[tid] -= S.fac[tid] * S.wv[p];
    __syncthreads();
  }
  if (tid == 0) {
    for (int i = 24; i >= 0; --i) {
      float acc = S.wv[i];
      for (int j = i + 1; j < 25; ++j) acc -= S.Wk[i * 25 + j] * S.wv[j];
      S.wv[i] = acc / S.Wk[i * 26];
    }
  }
  __syncthreads();
  // dz = -(t_l - G_l dy') ; dy = -dy'
  for (int idx = tid; idx < 125; idx += nthr) {
    int j = idx / 5, l = idx % 5;
    float acc = S.tv[l * 25 + j];
    for (int f = 0; f < 25; ++f) acc -= S.Gm[l][j * 25 + f] * S.wv[f];
    dz[idx] = -acc;
  }
  if (tid < 25) dy[tid] = -S.wv[tid];
  __syncthreads();
  // dx = Q^-1 (-rx - dz - A^T dy) ; ds = (-rs - dz)/d
  for (int idx = tid; idx < 125; idx += nthr) {
    int i = idx / 5, k = idx % 5;
    float acc = 0.f;
    for (int j = 0; j < 25; ++j) {
      float g1 = -rx[j * 5 + k] - dz[j * 5 + k] - dy[j];
      acc += S.Minv[i * 25 + j] * g1;
    }
    dx[idx] = acc;
    ds[idx] = (-rs[idx] - dz[idx]) / d[idx];
  }
  __syncthreads();
}

__global__ void __launch_bounds__(128) k_qp(const float* __restrict__ Kb,
                                            float* __restrict__ qp, int B) {
  __shared__ QPS S;
  int b = blockIdx.x, tid = threadIdx.x, nthr = blockDim.x;
  const float* Kg = Kb + (size_t)b * NSP * NSP;
  for (int idx = tid; idx < 625; idx += nthr) {
    int i = idx / 25, j = idx % 25;
    float v = Kg[i * NSP + j] + ((i == j) ? 1.f : 0.f);   // M = K + I
    S.Mm[idx] = v;
    S.Lm[idx] = v;
  }
  for (int idx = tid; idx < 125; idx += nthr) S.zer125[idx] = 0.f;
  if (tid < 25) S.zer25[tid] = 0.f;
  __syncthreads();
  chol_multi(S.Lm, 1, tid, nthr);
  // Minv via 25 RHS = I
  if (tid < 25) {
    float v[25];
    for (int i = 0; i < 25; ++i) {
      float acc = (i == tid) ? 1.f : 0.f;
      for (int j = 0; j < i; ++j) acc -= S.Lm[i * 25 + j] * v[j];
      v[i] = acc / S.Lm[i * 26];
    }
    for (int i = 24; i >= 0; --i) {
      float acc = v[i];
      for (int j = i + 1; j < 25; ++j) acc -= S.Lm[j * 25 + i] * v[j];
      v[i] = acc / S.Lm[i * 26];
    }
    for (int i = 0; i < 25; ++i) S.Minv[i * 25 + tid] = v[i];
  }
  __syncthreads();
  // initial KKT: d=1, rx=p=-onehot, rs=0, rz=-h, ry=0
  for (int idx = tid; idx < 125; idx += nthr) {
    int i = idx / 5, k = idx % 5;
    float oh = ((i % 5) == k) ? 1.f : 0.f;
    S.dvec[idx] = 1.f;
    S.rxv[idx] = -oh;
    S.rzv[idx] = -C_REG * oh;
  }
  __syncthreads();
  solve_kkt(S, S.dvec, S.rxv, S.zer125, S.rzv, S.zer25,
            S.xv, S.sv, S.zv, S.yv, true, tid, nthr);
  // shift s, z
  if (tid == 0) {
    float ms = 1e30f, mz = 1e30f;
    for (int i = 0; i < 125; ++i) { ms = fminf(ms, S.sv[i]); mz = fminf(mz, S.zv[i]); }
    S.red[0] = (ms < 0.f) ? (ms - 1.f) : 0.f;
    S.red[1] = (mz < 0.f) ? (mz - 1.f) : 0.f;
    S.red[2] = 1e30f;                       // best_res
  }
  __syncthreads();
  for (int idx = tid; idx < 125; idx += nthr) {
    S.sv[idx] -= S.red[0];
    S.zv[idx] -= S.red[1];
    S.bestx[idx] = S.xv[idx];
  }
  __syncthreads();
  for (int it = 0; it < 3; ++it) {
    // residuals
    for (int idx = tid; idx < 125; idx += nthr) {
      int i = idx / 5, k = idx % 5;
      float oh = ((i % 5) == k) ? 1.f : 0.f;
      float acc = S.yv[i] + S.zv[idx] - oh;   // A^T y + z + p
      for (int j = 0; j < 25; ++j) acc += S.Mm[i * 25 + j] * S.xv[j * 5 + k]; // + Q x
      S.rxv[idx] = acc;
      S.rsv[idx] = S.zv[idx];
      S.rzv[idx] = S.xv[idx] + S.sv[idx] - C_REG * oh;
    }
    if (tid < 25) {
      float acc = 0.f;
      for (int k = 0; k < 5; ++k) acc += S.xv[tid * 5 + k];
      S.ryv[tid] = acc;
    }
    __syncthreads();
    if (tid == 0) {
      float sz = 0.f, nrx = 0.f, nrz = 0.f, nry = 0.f;
      for (int i = 0; i < 125; ++i) {
        sz += S.sv[i] * S.zv[i];
        nrx += S.rxv[i] * S.rxv[i];
        nrz += S.rzv[i] * S.rzv[i];
      }
      for (int i = 0; i < 25; ++i) nry += S.ryv[i] * S.ryv[i];
      float mu = fabsf(sz) / 125.f;
      float res = sqrtf(nrz + 1e-30f) + sqrtf(nry + 1e-30f) + sqrtf(nrx + 1e-30f)
                + 125.f * mu;
      S.red[3] = mu;
      S.red[4] = (res < S.red[2]) ? 1.f : 0.f;
      if (res < S.red[2]) S.red[2] = res;
      S.red[5] = sz;
    }
    __syncthreads();
    if (S.red[4] != 0.f) {
      for (int idx = tid; idx < 125; idx += nthr) S.bestx[idx] = S.xv[idx];
    }
    __syncthreads();
    if (it == 2) break;
    for (int idx = tid; idx < 125; idx += nthr) S.dvec[idx] = S.zv[idx] / S.sv[idx];
    __syncthreads();
    // affine step
    solve_kkt(S, S.dvec, S.rxv, S.rsv, S.rzv, S.ryv,
              S.dxa, S.dsa, S.dza, S.dya, true, tid, nthr);
    if (tid == 0) {
      float a1 = fminf(fminf(step_min125(S.zv, S.dza), step_min125(S.sv, S.dsa)), 1.f);
      float num = 0.f;
      for (int i = 0; i < 125; ++i)
        num += (S.sv[i] + a1 * S.dsa[i]) * (S.zv[i] + a1 * S.dza[i]);
      float ratio = num / S.red[5];
      S.red[6] = -S.red[3] * ratio * ratio * ratio;   // -mu*sigma
    }
    __syncthreads();
    for (int idx = tid; idx < 125; idx += nthr)
      S.rsc[idx] = (S.red[6] + S.dsa[idx] * S.dza[idx]) / S.sv[idx];
    __syncthreads();
    // corrector (reuse factorization: same d)
    solve_kkt(S, S.dvec, S.zer125, S.rsc, S.zer125, S.zer25,
              S.dxc, S.dsc, S.dzc, S.dyc, false, tid, nthr);
    for (int idx = tid; idx < 125; idx += nthr) {
      S.dxa[idx] += S.dxc[idx];
      S.dsa[idx] += S.dsc[idx];
      S.dza[idx] += S.dzc[idx];
    }
    if (tid < 25) S.dya[tid] += S.dyc[tid];
    __syncthreads();
    if (tid == 0) {
      float a2 = fminf(0.999f * fminf(step_min125(S.zv, S.dza), step_min125(S.sv, S.dsa)),
                       1.f);
      S.red[7] = a2;
    }
    __syncthreads();
    float a2 = S.red[7];
    for (int idx = tid; idx < 125; idx += nthr) {
      S.xv[idx] += a2 * S.dxa[idx];
      S.sv[idx] += a2 * S.dsa[idx];
      S.zv[idx] += a2 * S.dza[idx];
    }
    if (tid < 25) S.yv[tid] += a2 * S.dya[tid];
    __syncthreads();
  }
  for (int idx = tid; idx < 125; idx += nthr) qp[(size_t)b * 125 + idx] = S.bestx[idx];
}

// ---------------- Kernel 5: out[b][q][w] = sum_s qp[b][s,w] * compat[b][s][q] ----------------
__global__ void __launch_bounds__(128) k_out(const float* __restrict__ qp,
                                             const float* __restrict__ compat,
                                             float* __restrict__ out, int B) {
  __shared__ float q[125];
  int b = blockIdx.x, tid = threadIdx.x;
  const float* qb = qp + (size_t)b * 125;
  const float* cb = compat + (size_t)b * NSP * NQP;
  if (tid < 125) q[tid] = qb[tid];
  __syncthreads();
  for (int idx = tid; idx < NQ * NWAY; idx += blockDim.x) {
    int qq = idx / NWAY, w = idx % NWAY;
    float acc = 0.f;
    for (int s = 0; s < NS; ++s) acc += q[s * NWAY + w] * cb[s * NQP + qq];
    out[(size_t)b * NQ * NWAY + idx] = acc;
  }
}

// ---------------- launch ----------------
extern "C" void kernel_launch(void* const* d_in, const int* in_sizes, int n_in,
                              void* d_out, int out_size, void* d_ws, size_t ws_size,
                              hipStream_t stream) {
  (void)n_in; (void)out_size; (void)ws_size;
  const float* ftest  = (const float*)d_in[0];
  const float* ftrain = (const float*)d_in[1];
  int B = in_sizes[1] / (NS * DF);

  char* ws = (char*)d_ws;
  size_t off = 0;
  auto alloc = [&](size_t bytes) -> void* {
    void* p = ws + off;
    off = (off + bytes + 255) & ~(size_t)255;
    return p;
  };
  _Float16* fqn    = (_Float16*)alloc((size_t)B * NQP * DF * sizeof(_Float16));
  _Float16* ftn    = (_Float16*)alloc((size_t)B * NSP * DF * sizeof(_Float16));
  float*    Kb     = (float*)alloc((size_t)B * NSP * NSP * sizeof(float));
  float*    compat = (float*)alloc((size_t)B * NSP * NQP * sizeof(float));
  float*    qp     = (float*)alloc((size_t)B * NZ * sizeof(float));
  float*    out    = (float*)d_out;

  k_prep  <<<B * (NQP + NSP), 128, 0, stream>>>(ftest, ftrain, fqn, ftn, B);
  k_gram  <<<B, 128, 0, stream>>>(ftn, Kb, B);
  k_compat<<<B, 320, 0, stream>>>(ftn, fqn, compat, B);
  k_qp    <<<B, 128, 0, stream>>>(Kb, qp, B);
  k_out   <<<B, 128, 0, stream>>>(qp, compat, out, B);
}
